// MultiHeadAttention_3238405342034
// MI455X (gfx1250) — compile-verified, matrix operand order checked by reading, not run on hardware
//
#include <hip/hip_runtime.h>
#include <hip/hip_bf16.h>

typedef _Float16 half_t;
typedef __attribute__((ext_vector_type(16))) _Float16 v16h;
typedef __attribute__((ext_vector_type(8)))  _Float16 v8h;
typedef __attribute__((ext_vector_type(4)))  _Float16 v4h;
typedef __attribute__((ext_vector_type(8)))  float    v8f;
typedef __attribute__((ext_vector_type(4)))  float    v4f;

#define DEVINL __device__ __forceinline__

// ---------------------------------------------------------------------------
// Fragment load: 16-bit A/B layout for v_wmma_*_16x16x32 (wave32).
// lane L holds row (L&15); element e holds K = ((e>>3)<<4) + ((L>>4)<<3) + (e&7)
// => two contiguous 16-byte chunks per lane at K-offsets {0, 16} (+8 if L>=16).
// For the B operand, pass a pointer into Bt where Bt[n][k] = B[k][n].
// ---------------------------------------------------------------------------
DEVINL v16h load_frag(const half_t* __restrict__ base, int ld) {
  const int lane = threadIdx.x & 31;
  const int r  = lane & 15;
  const int kb = (lane >> 4) << 3;            // 0 or 8
  const half_t* p = base + (size_t)r * ld + kb;
  v8h lo = *(const v8h*)(p);
  v8h hi = *(const v8h*)(p + 16);
  return __builtin_shufflevector(lo, hi, 0,1,2,3,4,5,6,7,8,9,10,11,12,13,14,15);
}

DEVINL v8f wmma_f16(v16h a, v16h b, v8f c) {
  return __builtin_amdgcn_wmma_f32_16x16x32_f16(false, a, false, b, (short)0, c,
                                                false, false);
}

// ---------------------------------------------------------------------------
// Precision conversion / packing kernels
// ---------------------------------------------------------------------------
__global__ void cvt_f32_f16(const float* __restrict__ in, half_t* __restrict__ out, int n4) {
  int i = blockIdx.x * blockDim.x + threadIdx.x;
  if (i < n4) {
    v4f v = ((const v4f*)in)[i];
    v4h o = { (half_t)v.x, (half_t)v.y, (half_t)v.z, (half_t)v.w };
    ((v4h*)out)[i] = o;
  }
}

// w: [H=16][C=1024][D=64] fp32  ->  wt: [n=h*64+d][c] f16 (Bt layout)
__global__ void pack_headw(const float* __restrict__ w, half_t* __restrict__ wt) {
  int idx = blockIdx.x * blockDim.x + threadIdx.x;   // 1M
  int c = idx & 1023;
  int n = idx >> 10;
  int h = n >> 6, d = n & 63;
  wt[idx] = (half_t)w[((size_t)((h << 10) + c) << 6) + d];
}

// w: [K=1024][N=1024] fp32 -> wt: [n][k] f16
__global__ void pack_proj(const float* __restrict__ w, half_t* __restrict__ wt) {
  int idx = blockIdx.x * blockDim.x + threadIdx.x;   // 1M
  int k = idx & 1023;
  int n = idx >> 10;
  wt[idx] = (half_t)w[((size_t)k << 10) + n];
}

// ---------------------------------------------------------------------------
// f16 GEMM: C[M,N] = A[M,K] * Bt[N,K]^T.  WG = 256 threads (8 waves),
// WG tile 128x128, wave tile 32x64 (2 m-tiles x 4 n-tiles -> 8 WMMA / k-step).
// MODE 0: store f16 row-major       (Q, K)
// MODE 1: store f16 head-transposed (V^T: [(m>>11)*1024 + n][m&2047])
// MODE 2: store fp32 + bias[n]      (output projection)
// ---------------------------------------------------------------------------
template <int MODE>
__global__ __launch_bounds__(256) void gemm_f16(
    const half_t* __restrict__ A, const half_t* __restrict__ Bt,
    void* __restrict__ Cout, const float* __restrict__ bias,
    int M, int N, int K)
{
  const int wave = threadIdx.x >> 5;
  const int lane = threadIdx.x & 31;
  const int m0 = blockIdx.x * 128 + (wave & 3) * 32;
  const int n0 = blockIdx.y * 128 + (wave >> 2) * 64;

  const half_t* aBase = A  + (size_t)m0 * K;
  const half_t* bBase = Bt + (size_t)n0 * K;

  v8f acc[2][4] = {};

  for (int kk = 0; kk < K; kk += 32) {
    __builtin_prefetch(aBase + kk + 128, 0, 1);   // global_prefetch_b8
    v16h a0 = load_frag(aBase + kk, K);
    v16h a1 = load_frag(aBase + (size_t)16 * K + kk, K);
#pragma unroll
    for (int nt = 0; nt < 4; ++nt) {
      v16h b = load_frag(bBase + (size_t)(nt * 16) * K + kk, K);
      acc[0][nt] = wmma_f16(a0, b, acc[0][nt]);
      acc[1][nt] = wmma_f16(a1, b, acc[1][nt]);
    }
  }

  // C/D layout: lane holds col n = lane&15; VGPR i holds row i + 8*(lane>=16)
  const int cn = lane & 15;
  const int mo = (lane >> 4) << 3;
#pragma unroll
  for (int mt = 0; mt < 2; ++mt) {
#pragma unroll
    for (int nt = 0; nt < 4; ++nt) {
#pragma unroll
      for (int i = 0; i < 8; ++i) {
        int m = m0 + mt * 16 + mo + i;
        int n = n0 + nt * 16 + cn;
        float val = acc[mt][nt][i];
        if (MODE == 0) {
          ((half_t*)Cout)[(size_t)m * N + n] = (half_t)val;
        } else if (MODE == 1) {
          ((half_t*)Cout)[((size_t)((m >> 11) * 1024 + n) << 11) + (m & 2047)] = (half_t)val;
        } else {
          ((float*)Cout)[(size_t)m * N + n] = val + bias[n];
        }
      }
    }
  }
}

// ---------------------------------------------------------------------------
// Flash attention (causal), per (b,h). Grid (T/64, B*H), 128 threads (4 waves).
// Each wave owns one 16-row query tile; streams 64-key blocks up to diagonal.
// Softmax runs in the exp2 domain: the C^-0.5*log2(e) scale is folded into the
// Q fragments once, so S comes out of WMMA pre-scaled. Row sums are computed
// with WMMA against an all-ones B fragment (no cross-lane shuffles needed).
// ---------------------------------------------------------------------------
__global__ __launch_bounds__(128) void attn_kernel(
    const half_t* __restrict__ Q, const half_t* __restrict__ Kmat,
    const half_t* __restrict__ Vt, half_t* __restrict__ Y)
{
  constexpr int T = 2048, C = 1024, D = 64, H = 16;
  const int bx   = blockIdx.x;
  const int bh   = blockIdx.y;
  const int b    = bh >> 4;
  const int h    = bh & (H - 1);
  const int wave = threadIdx.x >> 5;
  const int lane = threadIdx.x & 31;
  const int qt0  = bx * 64 + wave * 16;

  const half_t* Qh  = Q    + (size_t)b * T * C + h * D;
  const half_t* Kh  = Kmat + (size_t)b * T * C + h * D;
  const half_t* Vth = Vt   + ((size_t)b * C + h * D) * T;  // [d][t], ld = T

  __shared__ half_t lds[4 * 16 * 72];          // per-wave 16x64 P tile (+pad)
  half_t* pbuf = lds + wave * (16 * 72);

  // Q fragments, pre-scaled by C^-0.5 * log2(e) so softmax runs in exp2 domain
  const half_t s2 = (half_t)(0.03125f * 1.44269504088896f);
  v16h qf0 = load_frag(Qh + (size_t)qt0 * C, C) * s2;
  v16h qf1 = load_frag(Qh + (size_t)qt0 * C + 32, C) * s2;

  // all-ones B fragment for WMMA row sums
  v16h ones;
#pragma unroll
  for (int i = 0; i < 16; ++i) ones[i] = (half_t)1.0f;

  v8f o[4] = {};
  float mrun[8], lrun[8];
#pragma unroll
  for (int i = 0; i < 8; ++i) { mrun[i] = -__builtin_inff(); lrun[i] = 0.f; }

  const int cn     = lane & 15;
  const int mo     = (lane >> 4) << 3;
  const int diagKb = bx * 64;

  for (int kb = 0; kb <= diagKb; kb += 64) {
    // ---- S = (Q*scale) K^T for this 16x64 block (already exp2-domain) --
    v8f s[4];
#pragma unroll
    for (int nt = 0; nt < 4; ++nt) {
      v16h b0 = load_frag(Kh + (size_t)(kb + nt * 16) * C, C);
      v16h b1 = load_frag(Kh + (size_t)(kb + nt * 16) * C + 32, C);
      v8f z = {};
      s[nt] = wmma_f16(qf0, b0, z);
      s[nt] = wmma_f16(qf1, b1, s[nt]);
    }
    // ---- causal mask on diagonal block --------------------------------
    if (kb == diagKb) {
#pragma unroll
      for (int nt = 0; nt < 4; ++nt)
#pragma unroll
        for (int i = 0; i < 8; ++i) {
          int scol = kb + nt * 16 + cn;
          int trow = qt0 + mo + i;
          if (scol > trow) s[nt][i] = -__builtin_inff();
        }
    }
    // ---- online max (16-lane shuffle reduction, rows live per index i) -
    float alpha[8];
#pragma unroll
    for (int i = 0; i < 8; ++i) {
      float v = fmaxf(fmaxf(s[0][i], s[1][i]), fmaxf(s[2][i], s[3][i]));
      v = fmaxf(v, __shfl_xor(v, 1));
      v = fmaxf(v, __shfl_xor(v, 2));
      v = fmaxf(v, __shfl_xor(v, 4));
      v = fmaxf(v, __shfl_xor(v, 8));
      float mn = fmaxf(mrun[i], v);
      alpha[i] = __builtin_amdgcn_exp2f(mrun[i] - mn);
      mrun[i]  = mn;
    }
    // ---- P = exp2(S - m); write C-layout -> LDS (A-layout reload) ------
#pragma unroll
    for (int nt = 0; nt < 4; ++nt)
#pragma unroll
      for (int i = 0; i < 8; ++i)
        pbuf[(mo + i) * 72 + nt * 16 + cn] =
            (half_t)__builtin_amdgcn_exp2f(s[nt][i] - mrun[i]);
    // ---- rescale O by alpha (overlaps with LDS traffic) ----------------
#pragma unroll
    for (int nt = 0; nt < 4; ++nt)
#pragma unroll
      for (int i = 0; i < 8; ++i) o[nt][i] *= alpha[i];

    __syncthreads();   // all waves share identical trip count; fences LDS
    v16h pf0 = load_frag(pbuf, 72);
    v16h pf1 = load_frag(pbuf + 32, 72);

    // ---- row sums via WMMA against ones (every column = rowsum) --------
    {
      v8f z = {};
      v8f rs = wmma_f16(pf0, ones, z);
      rs = wmma_f16(pf1, ones, rs);
#pragma unroll
      for (int i = 0; i < 8; ++i) lrun[i] = lrun[i] * alpha[i] + rs[i];
    }

    // ---- O += P V  (B-frag from V^T, ld = T) ---------------------------
#pragma unroll
    for (int nt = 0; nt < 4; ++nt) {
      v16h vb0 = load_frag(Vth + (size_t)(nt * 16) * T + kb, T);
      v16h vb1 = load_frag(Vth + (size_t)(nt * 16) * T + kb + 32, T);
      o[nt] = wmma_f16(pf0, vb0, o[nt]);
      o[nt] = wmma_f16(pf1, vb1, o[nt]);
    }
    __syncthreads();   // protect pbuf reuse next iteration
  }

  // ---- epilogue: Y[b*T+t][h*64+d] = O / l  (f16 for next GEMM) ----------
#pragma unroll
  for (int i = 0; i < 8; ++i) {
    float inv = 1.f / lrun[i];
    int   t   = qt0 + mo + i;
    size_t row = ((size_t)b * T + t) * C + h * D;
#pragma unroll
    for (int nt = 0; nt < 4; ++nt)
      Y[row + nt * 16 + cn] = (half_t)(o[nt][i] * inv);
  }
}

// ---------------------------------------------------------------------------
// Host launcher
// ---------------------------------------------------------------------------
extern "C" void kernel_launch(void* const* d_in, const int* in_sizes, int n_in,
                              void* d_out, int out_size, void* d_ws, size_t ws_size,
                              hipStream_t stream) {
  const float* x  = (const float*)d_in[0];
  const float* wq = (const float*)d_in[1];
  const float* wk = (const float*)d_in[2];
  const float* wv = (const float*)d_in[3];
  const float* wp = (const float*)d_in[4];
  const float* bp = (const float*)d_in[5];
  float* out = (float*)d_out;

  constexpr int    B = 8, T = 2048, C = 1024, H = 16;
  constexpr size_t MT = (size_t)B * T;            // 16384 rows

  char* ws = (char*)d_ws;
  size_t off = 0;
  auto alloc = [&](size_t bytes) {
    char* p = ws + off;
    off += (bytes + 255) & ~(size_t)255;
    return p;
  };
  half_t* xh  = (half_t*)alloc(MT * C * 2);
  half_t* wqT = (half_t*)alloc((size_t)C * C * 2);
  half_t* wkT = (half_t*)alloc((size_t)C * C * 2);
  half_t* wvT = (half_t*)alloc((size_t)C * C * 2);
  half_t* wpT = (half_t*)alloc((size_t)C * C * 2);
  half_t* Qb  = (half_t*)alloc(MT * C * 2);
  half_t* Kb  = (half_t*)alloc(MT * C * 2);
  half_t* Vtb = (half_t*)alloc(MT * C * 2);
  half_t* Yb  = (half_t*)alloc(MT * C * 2);

  // 1) precision conversion + weight packing
  int n4 = (int)(MT * C / 4);
  cvt_f32_f16<<<(n4 + 255) / 256, 256, 0, stream>>>(x, xh, n4);
  pack_headw<<<(C * C) / 256, 256, 0, stream>>>(wq, wqT);
  pack_headw<<<(C * C) / 256, 256, 0, stream>>>(wk, wkT);
  pack_headw<<<(C * C) / 256, 256, 0, stream>>>(wv, wvT);
  pack_proj <<<(C * C) / 256, 256, 0, stream>>>(wp, wpT);

  // 2) QKV projections (WMMA GEMM)
  dim3 g1((unsigned)(MT / 128), (unsigned)(C / 128));
  gemm_f16<0><<<g1, 256, 0, stream>>>(xh, wqT, Qb,  nullptr, (int)MT, C, C);
  gemm_f16<0><<<g1, 256, 0, stream>>>(xh, wkT, Kb,  nullptr, (int)MT, C, C);
  gemm_f16<1><<<g1, 256, 0, stream>>>(xh, wvT, Vtb, nullptr, (int)MT, C, C);

  // 3) causal flash attention
  dim3 g2((unsigned)(T / 64), (unsigned)(B * H));
  attn_kernel<<<g2, 128, 0, stream>>>(Qb, Kb, Vtb, Yb);

  // 4) output projection + bias (fp32 out)
  gemm_f16<2><<<g1, 256, 0, stream>>>(Yb, wpT, out, bp, (int)MT, C, C);
}
